// se_invariant_net_19774029431055
// MI455X (gfx1250) — matrix operand order repaired and checked by reading, not compiled, not run on hardware
//
#include <hip/hip_runtime.h>
#include <hip/hip_bf16.h>

typedef __attribute__((ext_vector_type(16))) _Float16 v16h;
typedef __attribute__((ext_vector_type(8)))  float    v8f;

#define BATCH 2
#define NPTS  1024
#define HID   64
#define NVALS 3

// out[b,i,v] = sum_j (...) + N * bf[v]; seed the bf term here (also clears poison).
__global__ void init_out_kernel(const float* __restrict__ bf, float* __restrict__ out) {
    int idx = blockIdx.x * blockDim.x + threadIdx.x;
    if (idx < BATCH * NPTS * NVALS) {
        out[idx] = (float)NPTS * bf[idx % NVALS];
    }
}

__global__ __launch_bounds__(256) void se_net_wmma_kernel(
    const float* __restrict__ x,
    const float* __restrict__ ln_scale,
    const float* __restrict__ ln_offset,
    const float* __restrict__ w1,
    const float* __restrict__ b1,
    const float* __restrict__ w2,
    const float* __restrict__ b2,
    const float* __restrict__ wf,
    float* __restrict__ out)
{
    __shared__ float xs[NPTS * 3];
    __shared__ float w1s[HID], b1s[HID], b2s[HID];

    const int tid   = threadIdx.x;
    const int blk   = blockIdx.x;
    const int b     = blk >> 6;        // blk / (NPTS/16)
    const int itile = blk & 63;
    const int ibase = itile * 16;

    // Cooperative LDS fill: whole x[b] row-block + tiny vectors.
    for (int idx = tid; idx < NPTS * 3; idx += 256) xs[idx] = x[b * NPTS * 3 + idx];
    if (tid < HID) { w1s[tid] = w1[tid]; b1s[tid] = b1[tid]; b2s[tid] = b2[tid]; }
    __syncthreads();

    const int lane = tid & 31;
    const int wave = tid >> 5;
    const int col  = lane & 15;   // WMMA B/C column -> the i index within the tile
    const int hi   = lane >> 4;   // lane half

    const float ls = ln_scale[0];
    const float lo = ln_offset[0];

    // ---- Preload static A-operands (transposed weights) in WMMA A layout ----
    // A 16x32 f16 as packed here: lane holds M = lane%16; element e maps to
    // K = 16*hi + e (contiguous k order, matched by B).
    v16h aw2[4][2];   // W2^T: 4 output-tiles (M) x 2 K-chunks
    v16h awf[2];      // Wf^T padded to 16 rows
    #pragma unroll
    for (int q = 0; q < 2; ++q) {
        #pragma unroll
        for (int e = 0; e < 16; ++e) {
            int kA = 16 * hi + e;
            int K  = 32 * q + kA;
            #pragma unroll
            for (int mt = 0; mt < 4; ++mt) {
                aw2[mt][q][e] = (_Float16)w2[K * HID + 16 * mt + col];
            }
            awf[q][e] = (col < NVALS) ? (_Float16)wf[K * NVALS + col] : (_Float16)0.0f;
        }
    }

    const int   ig  = ibase + col;
    const float xi0 = xs[ig * 3 + 0];
    const float xi1 = xs[ig * 3 + 1];
    const float xi2 = xs[ig * 3 + 2];

    // Layer-3 accumulator: D[M=val, N=i], f32, carries the sum over j.
    v8f acc = {0.f, 0.f, 0.f, 0.f, 0.f, 0.f, 0.f, 0.f};

    const int j0 = wave * (NPTS / 8);
    const int j1 = j0 + (NPTS / 8);
    for (int j = j0; j < j1; ++j) {
        // pairwise distance, faithful to reference (diff + 1e-10, diagonal zeroed)
        float d0 = xs[j * 3 + 0] - xi0 + 1e-10f;
        float d1 = xs[j * 3 + 1] - xi1 + 1e-10f;
        float d2 = xs[j * 3 + 2] - xi2 + 1e-10f;
        float nrm = sqrtf(d0 * d0 + d1 * d1 + d2 * d2);
        nrm = (j == ig) ? 0.0f : nrm;

        // LayerNorm over a size-1 axis, computed generically.
        float hh  = nrm;
        float mu  = hh;
        float ctr = hh - mu;
        float var = ctr * ctr;
        float t   = ctr * rsqrtf(var + 1e-5f) * ls + lo;

        // h1^T in B-operand layout: element e of chunk q is K = 32q + 16*hi + e,
        // column = col. Built elementwise (1->64 layer is an outer product).
        v16h bh1[2];
        #pragma unroll
        for (int q = 0; q < 2; ++q) {
            #pragma unroll
            for (int e = 0; e < 16; ++e) {
                int k = 32 * q + 16 * hi + e;
                float h1 = fmaxf(fmaf(t, w1s[k], b1s[k]), 0.0f);
                bh1[q][e] = (_Float16)h1;
            }
        }

        // Layer 2: h2^T = W2^T @ h1^T  (8 wmma f32_16x16x32_f16)
        // D layout: lane holds column i; VGPR v holds hidden = 16*mt + v + 8*hi.
        float h2v[4][8];
        #pragma unroll
        for (int mt = 0; mt < 4; ++mt) {
            v8f c2 = {0.f, 0.f, 0.f, 0.f, 0.f, 0.f, 0.f, 0.f};
            c2 = __builtin_amdgcn_wmma_f32_16x16x32_f16(false, aw2[mt][0], false, bh1[0],
                                                        (short)0, c2, false, false);
            c2 = __builtin_amdgcn_wmma_f32_16x16x32_f16(false, aw2[mt][1], false, bh1[1],
                                                        (short)0, c2, false, false);
            #pragma unroll
            for (int v = 0; v < 8; ++v) {
                h2v[mt][v] = fmaxf(c2[v] + b2s[16 * mt + 8 * hi + v], 0.0f);
            }
        }

        // Repack h2^T from D layout into B-operand layout for layer 3.
        // Element e of chunk q needs K = 32q + e + 16*hi, held at
        // [mt = 2q + hi][v = e&7] on the lane half hs = (e>>3)&1 of the same
        // column -> at most one cross-half exchange per element.
        v16h bh2[2];
        #pragma unroll
        for (int q = 0; q < 2; ++q) {
            #pragma unroll
            for (int e = 0; e < 16; ++e) {
                int   vs   = e & 7;
                int   hs   = (e >> 3) & 1;
                float la   = h2v[2 * q + 0][vs];
                float lb   = h2v[2 * q + 1][vs];
                float mine = hi ? lb : la;                 // h2v[2q+hi][vs] (local)
                float send = hi ? la : lb;                 // what the partner half needs
                float remote = __shfl_xor(send, 16, 32);   // partner's h2v[2q+our_hi][vs]
                float val  = (hs == hi) ? mine : remote;
                bh2[q][e]  = (_Float16)val;
            }
        }

        // Layer 3: accumulate Wf^T @ h2^T into the persistent C register
        // (the sum over j lives entirely in the WMMA accumulator).
        acc = __builtin_amdgcn_wmma_f32_16x16x32_f16(false, awf[0], false, bh2[0],
                                                     (short)0, acc, false, false);
        acc = __builtin_amdgcn_wmma_f32_16x16x32_f16(false, awf[1], false, bh2[1],
                                                     (short)0, acc, false, false);
    }

    // D rows 0..2 (lanes 0-15, VGPRs 0-2) are out[b, ibase+col, 0..2] partial sums.
    if (hi == 0) {
        #pragma unroll
        for (int v = 0; v < NVALS; ++v) {
            atomicAdd(&out[(b * NPTS + ig) * NVALS + v], acc[v]);
        }
    }
}

extern "C" void kernel_launch(void* const* d_in, const int* in_sizes, int n_in,
                              void* d_out, int out_size, void* d_ws, size_t ws_size,
                              hipStream_t stream) {
    const float* x         = (const float*)d_in[0];
    const float* ln_scale  = (const float*)d_in[1];
    const float* ln_offset = (const float*)d_in[2];
    const float* w1        = (const float*)d_in[3];
    const float* b1        = (const float*)d_in[4];
    const float* w2        = (const float*)d_in[5];
    const float* b2        = (const float*)d_in[6];
    const float* wf        = (const float*)d_in[7];
    const float* bf        = (const float*)d_in[8];
    float* out = (float*)d_out;

    init_out_kernel<<<(BATCH * NPTS * NVALS + 255) / 256, 256, 0, stream>>>(bf, out);
    se_net_wmma_kernel<<<BATCH * (NPTS / 16), 256, 0, stream>>>(
        x, ln_scale, ln_offset, w1, b1, w2, b2, wf, out);
}
